// VO2LSNN_33938831573583
// MI455X (gfx1250) — matrix operand order repaired and checked by reading, not compiled, not run on hardware
//
#include <hip/hip_runtime.h>

typedef __attribute__((ext_vector_type(16))) _Float16 v16h;
typedef __attribute__((ext_vector_type(8)))  float    v8f;
typedef unsigned int u32;
typedef __attribute__((ext_vector_type(4))) u32 v4u;

#define T_STEPS 300
#define BATCH   32
#define NIN     700
#define NINP    704          // padded per-delay block (multiple of 32)
#define NH      256
#define NOUT    20
#define DMAX    10
#define KIN     (DMAX*NINP)  // 7040
#define KREC    (DMAX*NH)    // 2560
#define NWG     16
// LDS row strides padded +8 f16 (16B) to spread lanes across all 64 banks
#define RSTRIDE 264          // ring row stride (f16): 528B = 132 dw = 4 mod 64 banks
#define WSTRIDE 2568         // wrec row stride (f16): 5136B = 1284 dw = 4 mod 64 banks

// ---- workspace layout (bytes) ----
#define OFF_XB     0
#define SZ_XB      ((T_STEPS+DMAX)*BATCH*NINP*2)          // f16 padded x
#define OFF_WINT   (OFF_XB + SZ_XB)
#define SZ_WINT    (NH*KIN*2)                              // W_in^T f16 [o][k]
#define OFF_WRECT  (OFF_WINT + SZ_WINT)
#define SZ_WRECT   (NH*KREC*2)                             // W_rec^T f16 [o][k]
#define OFF_CURIN  (OFF_WRECT + SZ_WRECT)
#define SZ_CURIN   (T_STEPS*BATCH*NH*4)                    // f32 input currents
#define OFF_SPK    (OFF_CURIN + SZ_CURIN)
#define SZ_SPK     (DMAX*BATCH*(NH/2)*4)                   // u32 = 2x f16 spikes
#define OFF_RPART  (OFF_SPK + SZ_SPK)
#define SZ_RPART   (NWG*BATCH*NOUT*4)                      // readout partials
#define OFF_BAR    (OFF_RPART + SZ_RPART)

// ---------------- prep kernels: build f16 WMMA-friendly layouts ----------------
__global__ void prep_x(const float* __restrict__ x, _Float16* __restrict__ xb, int n){
  for (int idx = blockIdx.x*blockDim.x + threadIdx.x; idx < n; idx += gridDim.x*blockDim.x){
    int i = idx % NINP; int r = idx / NINP; int b = r % BATCH; int tt = r / BATCH;
    int t = tt - DMAX;                       // 10 leading zero timesteps for delays
    float v = 0.f;
    if (t >= 0 && i < NIN) v = x[(t*BATCH + b)*NIN + i];
    xb[idx] = (_Float16)v;
  }
}

__global__ void prep_win(const float* __restrict__ w_in, _Float16* __restrict__ winT, int n){
  for (int idx = blockIdx.x*blockDim.x + threadIdx.x; idx < n; idx += gridDim.x*blockDim.x){
    int o = idx / KIN; int k = idx % KIN; int d = k / NINP; int i = k % NINP;
    float v = 0.f;
    if (i < NIN) v = w_in[(i*NH + o)*DMAX + d];
    winT[idx] = (_Float16)v;
  }
}

__global__ void prep_wrec(const float* __restrict__ w_rec, _Float16* __restrict__ wrecT,
                          unsigned* __restrict__ bar, int n){
  if (blockIdx.x == 0 && threadIdx.x == 0) bar[0] = 0u;   // deterministic barrier reset
  for (int idx = blockIdx.x*blockDim.x + threadIdx.x; idx < n; idx += gridDim.x*blockDim.x){
    int o = idx / KREC; int k = idx % KREC; int d = k / NH; int j = k % NH;
    wrecT[idx] = (_Float16)w_rec[(j*NH + o)*DMAX + d];
  }
}

// ---------------- phase 1: cur_in[t,b,o] = sum_d x[t-d] @ W_in[:,:,d] ----------------
// grid = 300 (one block per t), block = 256 (8 waves). Wave w owns n-tiles {w, w+8},
// both batch halves: 4 independent accumulators, K = 7040 (220 k-iters of 32).
// Fragment loads: 2x global_load_b128 per fragment, strength-reduced base pointers.
__global__ void __launch_bounds__(256) phase1(const _Float16* __restrict__ xb,
                                              const _Float16* __restrict__ winT,
                                              float* __restrict__ curin){
  const int t  = blockIdx.x;
  const int w  = threadIdx.x >> 5;
  const int L  = threadIdx.x & 31;
  const int lh = L >> 4, lm = L & 15;
  const v4u* xq = (const v4u*)xb;
  const v4u* wq = (const v4u*)winT;
  v8f a00{}, a01{}, a10{}, a11{};
  const int nt0 = w, nt1 = w + 8;
  const int b0 = lm, b1 = 16 + lm;
  const int o0 = nt0*16 + lm, o1 = nt1*16 + lm;
  const v4u* qb0 = wq + o0*(KIN/8) + lh;
  const v4u* qb1 = wq + o1*(KIN/8) + lh;
  for (int d = 0; d < DMAX; ++d){
    const int tt = t + DMAX - d;
    const v4u* qa0 = xq + (tt*BATCH + b0)*(NINP/8) + lh;
    const v4u* qa1 = xq + (tt*BATCH + b1)*(NINP/8) + lh;
    const v4u* qd0 = qb0 + d*(NINP/8);
    const v4u* qd1 = qb1 + d*(NINP/8);
    for (int q = 0; q < NINP/32; ++q){
      union { v16h h; v4u v[2]; } A0, A1, B0, B1;
      A0.v[0] = qa0[q*4];  A0.v[1] = qa0[q*4 + 2];
      A1.v[0] = qa1[q*4];  A1.v[1] = qa1[q*4 + 2];
      B0.v[0] = qd0[q*4];  B0.v[1] = qd0[q*4 + 2];
      B1.v[0] = qd1[q*4];  B1.v[1] = qd1[q*4 + 2];
      a00 = __builtin_amdgcn_wmma_f32_16x16x32_f16(false, A0.h, false, B0.h, (short)0, a00, false, false);
      a01 = __builtin_amdgcn_wmma_f32_16x16x32_f16(false, A0.h, false, B1.h, (short)0, a01, false, false);
      a10 = __builtin_amdgcn_wmma_f32_16x16x32_f16(false, A1.h, false, B0.h, (short)0, a10, false, false);
      a11 = __builtin_amdgcn_wmma_f32_16x16x32_f16(false, A1.h, false, B1.h, (short)0, a11, false, false);
    }
  }
  #pragma unroll
  for (int r = 0; r < 8; ++r){
    const int m0 = r + (lh << 3), m1 = 16 + m0;    // C-frag row map
    curin[(t*BATCH + m0)*NH + o0] = a00[r];
    curin[(t*BATCH + m0)*NH + o1] = a01[r];
    curin[(t*BATCH + m1)*NH + o0] = a10[r];
    curin[(t*BATCH + m1)*NH + o1] = a11[r];
  }
}

// ---------------- helpers ----------------
__device__ __forceinline__ float absid(float kwl, float vgs, float vth, float vds){
  float vov = fmaxf(vgs - vth, 0.f);
  float vda = fabsf(vds);
  float tri = (vov - 0.5f*vda)*vda;
  float sat = 0.5f*vov*vov;
  float mag = (vda < vov) ? tri : sat;
  float sgn = (vds > 0.f) ? 1.f : ((vds < 0.f) ? -1.f : 0.f);
  return kwl * sgn * mag;
}

__device__ __forceinline__ void cluster_sync(unsigned* bar, unsigned tgt, int tid){
  __syncthreads();
  if (tid == 0)
    __hip_atomic_fetch_add(bar, 1u, __ATOMIC_RELEASE, __HIP_MEMORY_SCOPE_AGENT);
  // CDNA5 cluster split-barrier fast path: one wave signals, all waves wait.
  // NOPs (per ISA) if the grid was not dispatched as a cluster; the atomic
  // generation barrier below then provides correctness.
  if ((tid >> 5) == 0) asm volatile("s_barrier_signal -3" ::: "memory");
  asm volatile("s_barrier_wait -3" ::: "memory");
  if (tid == 0){
    while (__hip_atomic_load(bar, __ATOMIC_ACQUIRE, __HIP_MEMORY_SCOPE_AGENT) < tgt)
      __builtin_amdgcn_s_sleep(1);
  }
  __syncthreads();
}

// ---------------- phase 2: persistent 16-WG cluster, 300 sequential steps ----------------
// Each WG owns 16 output neurons x 32 batch. LDS: W_rec^T slice (resident all 300
// steps) + 10-slot full spike-history ring, both bank-conflict padded. Per step:
// 2x80 WMMAs for cur_rec (K=2560 over delayed spikes) in two pipelined acc chains.
__global__ void __launch_bounds__(64) phase2(const float* __restrict__ curin,
                                             const _Float16* __restrict__ wrecT,
                                             const float* __restrict__ w_out,
                                             const float* __restrict__ b_out,
                                             u32* __restrict__ spk_g,
                                             float* __restrict__ rparts,
                                             unsigned* __restrict__ bar,
                                             float* __restrict__ out){
  __shared__ _Float16 wrec_s[16*WSTRIDE];         // ~80 KB (padded rows)
  __shared__ _Float16 ring[DMAX*BATCH*RSTRIDE];   // ~165 KB (padded rows)
  __shared__ float cur_s[BATCH*16];
  __shared__ float sstage[BATCH*16];
  __shared__ float wout_s[16*NOUT];
  __shared__ float y_s[BATCH*NOUT];

  const int wg  = blockIdx.x;
  const int tid = threadIdx.x;
  const int L   = tid & 31;
  const int wv  = tid >> 5;                       // wave id = batch-half tile
  const int lh  = L >> 4, lm = L & 15;

  { // init: LDS-resident weight slice (padded rows), zero ring, readout weights
    const v4u* src = (const v4u*)(wrecT + wg*16*KREC);
    v4u* dst = (v4u*)wrec_s;
    for (int i = tid; i < 16*(KREC/8); i += 64){
      const int row = i / (KREC/8), col = i % (KREC/8);
      dst[row*(WSTRIDE/8) + col] = src[i];
    }
    v4u* rz = (v4u*)ring;
    const v4u zz = (v4u)(0u);
    for (int i = tid; i < DMAX*BATCH*RSTRIDE/8; i += 64) rz[i] = zz;
    for (int i = tid; i < 16*NOUT; i += 64) wout_s[i] = w_out[(wg*16)*NOUT + i];
    if (wg == 0) for (int i = tid; i < BATCH*NOUT; i += 64) y_s[i] = 0.f;
  }
  __syncthreads();

  const float FCT = __expf(-0.001f/(2e-7f*101500.f));  // exp(-DT/(CMEM*(RH+RS)))
  const float FVA = __expf(-0.001f/(7e-6f*100000.f));  // exp(-DT/(CA*RA))
  const float ALP = __expf(-0.001f/0.02f);             // exp(-DT/LP_TAU)

  float nv[8], nva[8], ncnt[8], nsp[8];
  #pragma unroll
  for (int e = 0; e < 8; ++e){ nv[e]=0.f; nva[e]=0.f; ncnt[e]=0.f; nsp[e]=0.f; }

  const v4u* ring_q = (const v4u*)ring;
  const v4u* wrec_q = (const v4u*)wrec_s + lm*(WSTRIDE/8) + lh;
  unsigned bar_tgt = 0;
  int slot = 0;
  const int bA = wv*16 + lm;                      // A-frag batch row

  for (int t = 0; t < T_STEPS; ++t){
    const int s0 = (slot == 0) ? (DMAX-1) : (slot-1);   // ring slot of step t-1
    // ---- cur = cur_in tile + spike-history @ W_rec; two pipelined acc chains ----
    v8f acc0, acc1{};
    const float* cip = curin + (t*BATCH + wv*16 + (lh << 3))*NH + wg*16 + lm;
    #pragma unroll
    for (int r = 0; r < 8; ++r) acc0[r] = cip[r*NH];
    for (int d = 0; d < DMAX; ++d){
      int sd = s0 - d; if (sd < 0) sd += DMAX;
      const v4u* aq = ring_q + (sd*BATCH + bA)*(RSTRIDE/8) + lh;
      const v4u* bq = wrec_q + d*(NH/8);
      if (d & 1){
        for (int q = 0; q < 8; ++q){
          union { v16h h; v4u v[2]; } A, B;
          A.v[0] = aq[q*4];  A.v[1] = aq[q*4 + 2];
          B.v[0] = bq[q*4];  B.v[1] = bq[q*4 + 2];
          acc1 = __builtin_amdgcn_wmma_f32_16x16x32_f16(false, A.h, false, B.h, (short)0, acc1, false, false);
        }
      } else {
        for (int q = 0; q < 8; ++q){
          union { v16h h; v4u v[2]; } A, B;
          A.v[0] = aq[q*4];  A.v[1] = aq[q*4 + 2];
          B.v[0] = bq[q*4];  B.v[1] = bq[q*4 + 2];
          acc0 = __builtin_amdgcn_wmma_f32_16x16x32_f16(false, A.h, false, B.h, (short)0, acc0, false, false);
        }
      }
    }
    #pragma unroll
    for (int r = 0; r < 8; ++r){
      const int b = wv*16 + r + (lh << 3);
      cur_s[b*16 + lm] = acc0[r] + acc1[r];
    }
    __syncthreads();

    // ---- neuron update: 8 neurons/thread, state in registers ----
    #pragma unroll
    for (int e = 0; e < 8; ++e){
      const int n = tid*8 + e, b = n >> 4, ol = n & 15;
      const float sp   = nsp[e];
      const bool  isrf = ncnt[e] > 0.f;
      float v  = (1.f - sp)*nv[e] + sp*0.5f;                       // reset (VRESET)
      float Ia = absid(1.8e-5f*4.f, 5.f*sp, 0.973f, 5.f - nva[e]); // KP*WLP, VDD, VTP
      float va = FVA*nva[e] + (1.f - FVA)*100000.f*Ia;             // RA
      float xin = fmaxf(cur_s[b*16 + ol], 0.f);
      float Il  = absid(2.9e-5f*6.f, va, 0.745f, v);               // KN*WLN, VTN
      float vnw = FCT*v + (1.f - FCT)*101500.f*(2.5e-4f*xin - Il);
      v = isrf ? v : vnw;
      float s = (v >= 2.0f) ? 1.f : 0.f;                           // VTH
      s = isrf ? 0.f : s;
      ncnt[e] = (s > 0.f) ? 5.f : fmaxf(ncnt[e] - 1.f, 0.f);       // REFRAC countdown
      nv[e] = v; nva[e] = va; nsp[e] = s;
      sstage[b*16 + ol] = s;
    }
    // publish spikes (f16 pairs) to L2 exchange buffer with agent scope
    #pragma unroll
    for (int p = 0; p < 4; ++p){
      union { _Float16 h[2]; u32 u; } pu;
      pu.h[0] = (_Float16)nsp[2*p];
      pu.h[1] = (_Float16)nsp[2*p + 1];
      const int n = tid*8 + 2*p, b = n >> 4, og = wg*16 + (n & 15);
      __hip_atomic_store(&spk_g[(slot*BATCH + b)*(NH/2) + (og >> 1)], pu.u,
                         __ATOMIC_RELAXED, __HIP_MEMORY_SCOPE_AGENT);
    }
    __syncthreads();

    // ---- readout partials: r_part = s_local @ w_out_rows ----
    #pragma unroll
    for (int p = 0; p < 10; ++p){
      const int idx = tid*10 + p;               // 640 = 32*20
      const int b = idx / NOUT, ko = idx % NOUT;
      float r = 0.f;
      #pragma unroll
      for (int ol = 0; ol < 16; ++ol) r += sstage[b*16 + ol]*wout_s[ol*NOUT + ko];
      __hip_atomic_store(&rparts[(wg*BATCH + b)*NOUT + ko], r,
                         __ATOMIC_RELAXED, __HIP_MEMORY_SCOPE_AGENT);
    }
    __threadfence();
    bar_tgt += NWG;
    cluster_sync(bar, bar_tgt, tid);

    // ---- gather full spike vector into local (padded) ring; WG0 finalizes readout ----
    {
      u32* rw = (u32*)ring;
      const int base = slot*BATCH*(NH/2);
      for (int i = tid; i < BATCH*(NH/2); i += 64){
        const int b = i >> 7, j = i & 127;      // NH/2 = 128 u32 per row
        rw[(slot*BATCH + b)*(RSTRIDE/2) + j] =
            __hip_atomic_load(&spk_g[base + i], __ATOMIC_RELAXED, __HIP_MEMORY_SCOPE_AGENT);
      }
    }
    if (wg == 0){
      #pragma unroll
      for (int p = 0; p < 10; ++p){
        const int idx = tid*10 + p;
        const int b = idx / NOUT, ko = idx % NOUT;
        float sum = b_out[ko];
        for (int g = 0; g < NWG; ++g)
          sum += __hip_atomic_load(&rparts[(g*BATCH + b)*NOUT + ko],
                                   __ATOMIC_RELAXED, __HIP_MEMORY_SCOPE_AGENT);
        const float y = ALP*y_s[idx] + (1.f - ALP)*sum;
        y_s[idx] = y;
        out[(t*BATCH + b)*NOUT + ko] = y;
      }
    }
    bar_tgt += NWG;
    cluster_sync(bar, bar_tgt, tid);            // r_parts consumed; ring slot ready
    slot = (slot == DMAX-1) ? 0 : slot + 1;
  }
}

// ---------------- launch ----------------
extern "C" void kernel_launch(void* const* d_in, const int* in_sizes, int n_in,
                              void* d_out, int out_size, void* d_ws, size_t ws_size,
                              hipStream_t stream){
  (void)in_sizes; (void)n_in; (void)out_size; (void)ws_size;
  const float* x     = (const float*)d_in[0];
  const float* w_in  = (const float*)d_in[1];
  const float* w_rec = (const float*)d_in[2];
  const float* w_out = (const float*)d_in[3];
  const float* b_out = (const float*)d_in[4];
  char* ws = (char*)d_ws;
  _Float16* xb    = (_Float16*)(ws + OFF_XB);
  _Float16* winT  = (_Float16*)(ws + OFF_WINT);
  _Float16* wrecT = (_Float16*)(ws + OFF_WRECT);
  float*    curin = (float*)   (ws + OFF_CURIN);
  u32*      spk   = (u32*)     (ws + OFF_SPK);
  float*    rpart = (float*)   (ws + OFF_RPART);
  unsigned* bar   = (unsigned*)(ws + OFF_BAR);
  float*    out   = (float*)d_out;

  const int nxb = (T_STEPS+DMAX)*BATCH*NINP;
  prep_x  <<<(nxb + 255)/256, 256, 0, stream>>>(x, xb, nxb);
  const int nwi = NH*KIN;
  prep_win<<<(nwi + 255)/256, 256, 0, stream>>>(w_in, winT, nwi);
  const int nwr = NH*KREC;
  prep_wrec<<<(nwr + 255)/256, 256, 0, stream>>>(w_rec, wrecT, bar, nwr);
  phase1<<<T_STEPS, 256, 0, stream>>>(xb, winT, curin);
  phase2<<<NWG, 64, 0, stream>>>(curin, wrecT, w_out, b_out, spk, rpart, bar, out);
}
